// HopfieldAttention_87522843558249
// MI455X (gfx1250) — matrix operand order repaired and checked by reading, not compiled, not run on hardware
//
#include <hip/hip_runtime.h>
#include <hip/hip_bf16.h>

#define HH 8
#define LL 2048
#define SS 2048
#define DD 512

typedef __attribute__((ext_vector_type(4)))  __bf16 v4bf;
typedef __attribute__((ext_vector_type(8)))  __bf16 v8bf;
typedef __attribute__((ext_vector_type(16))) __bf16 v16bf;
typedef __attribute__((ext_vector_type(4)))  float  v4f;
typedef __attribute__((ext_vector_type(8)))  float  v8f;

// D = A(16x32 bf16) * B(32x16 bf16) + C(16x16 f32)
static __device__ __forceinline__ v8f wmma_bf16(v16bf a, v16bf b, v8f c) {
  return __builtin_amdgcn_wmma_f32_16x16x32_bf16(false, a, false, b,
                                                 (short)0, c, false, false);
}

// A-fragment: lane (m = lane&15, hi = lane>>4) holds K = hi*16+[0,16)
static __device__ __forceinline__ v16bf load_afrag(const __bf16* base, int stride,
                                                   int m, int hi) {
  v8bf lo = *(const v8bf*)(base + m * stride + hi * 16);
  v8bf hh = *(const v8bf*)(base + m * stride + hi * 16 + 8);
  return __builtin_shufflevector(lo, hh, 0,1,2,3,4,5,6,7,8,9,10,11,12,13,14,15);
}

// B-fragment from transposed storage Bt[n][k]: lane (n = lane&15) holds K = hi*16+[0,16)
static __device__ __forceinline__ v16bf load_bfrag(const __bf16* base, int stride,
                                                   int n, int hi) {
  return *(const v16bf*)(base + n * stride + hi * 16);
}

// CDNA5 async DMA: global -> LDS, 16 bytes per lane, tracked by ASYNCcnt.
// GVS mode: 64-bit SGPR base + 32-bit per-lane VGPR byte offset.
// lds_dst low 32 bits of a generic pointer into __shared__ == LDS byte offset.
static __device__ __forceinline__ void async_g2l_b128(const void* lds_dst,
                                                      const void* g_base,
                                                      unsigned byte_off) {
  unsigned ldsa = (unsigned)(unsigned long long)lds_dst;
  asm volatile("global_load_async_to_lds_b128 %0, %1, %2"
               :
               : "v"(ldsa), "v"(byte_off), "s"(g_base)
               : "memory");
}
static __device__ __forceinline__ void wait_async0() {
  asm volatile("s_wait_asynccnt 0x0" ::: "memory");
}

// ---------------------------------------------------------------------------
// QKV projection: Y(bf16, (B,H,T,64)) = X(f32, (M,512)) @ W^T + bias
// block = 128 threads (4 waves), 64x64 output tile
// ---------------------------------------------------------------------------
__global__ __launch_bounds__(128) void proj_kernel(
    const float* __restrict__ X, const float* __restrict__ W,
    const float* __restrict__ bias, __bf16* __restrict__ Y) {
  __shared__ __align__(32) __bf16 ldsA[64 * 32];
  __shared__ __align__(32) __bf16 ldsW[64 * 32];
  const int tid = threadIdx.x, wave = tid >> 5, lane = tid & 31;
  const int m16 = lane & 15, hi = lane >> 4;
  const int mb = blockIdx.x * 64, nb = blockIdx.y * 64;
  const v8f vz = {0.f, 0.f, 0.f, 0.f, 0.f, 0.f, 0.f, 0.f};
  v8f acc[4] = {vz, vz, vz, vz};

  for (int k0 = 0; k0 < DD; k0 += 32) {
    __syncthreads();
#pragma unroll
    for (int it = 0; it < 4; ++it) {
      int e = (tid + it * 128) * 4;
      int rl = e >> 5, kl = e & 31;
      v4f xa = *(const v4f*)(X + (size_t)(mb + rl) * DD + k0 + kl);
      v4bf xb = {(__bf16)xa.x, (__bf16)xa.y, (__bf16)xa.z, (__bf16)xa.w};
      *(v4bf*)(ldsA + rl * 32 + kl) = xb;
      v4f wa = *(const v4f*)(W + (size_t)(nb + rl) * DD + k0 + kl);
      v4bf wb = {(__bf16)wa.x, (__bf16)wa.y, (__bf16)wa.z, (__bf16)wa.w};
      *(v4bf*)(ldsW + rl * 32 + kl) = wb;
    }
    __syncthreads();
    v16bf af = load_afrag(ldsA, 32, wave * 16 + m16, hi);
#pragma unroll
    for (int j = 0; j < 4; ++j) {
      v16bf bf = load_bfrag(ldsW, 32, j * 16 + m16, hi);
      acc[j] = wmma_bf16(af, bf, acc[j]);
    }
  }
#pragma unroll
  for (int j = 0; j < 4; ++j) {
    int n_g = nb + j * 16 + m16;
    float bv = bias[n_g];
    int hd = n_g >> 6, dh = n_g & 63;
#pragma unroll
    for (int r = 0; r < 8; ++r) {
      int m_g = mb + wave * 16 + r + hi * 8;
      int bb = m_g >> 11, ll = m_g & (LL - 1);
      Y[((((size_t)bb * HH + hd) * LL + ll) << 6) + dh] = (__bf16)(acc[j][r] + bv);
    }
  }
}

// ---------------------------------------------------------------------------
// Output projection: Y(f32, (M,512)) = X(bf16, (M,512)) @ W^T + bias
// ---------------------------------------------------------------------------
__global__ __launch_bounds__(128) void oproj_kernel(
    const __bf16* __restrict__ X, const float* __restrict__ W,
    const float* __restrict__ bias, float* __restrict__ Y) {
  __shared__ __align__(32) __bf16 ldsA[64 * 32];
  __shared__ __align__(32) __bf16 ldsW[64 * 32];
  const int tid = threadIdx.x, wave = tid >> 5, lane = tid & 31;
  const int m16 = lane & 15, hi = lane >> 4;
  const int mb = blockIdx.x * 64, nb = blockIdx.y * 64;
  const v8f vz = {0.f, 0.f, 0.f, 0.f, 0.f, 0.f, 0.f, 0.f};
  v8f acc[4] = {vz, vz, vz, vz};

  for (int k0 = 0; k0 < DD; k0 += 32) {
    __syncthreads();
#pragma unroll
    for (int it = 0; it < 2; ++it) {
      int ch = tid + it * 128;             // 256 chunks of 8 halves
      int rl = ch >> 2, kl = (ch & 3) * 8;
      *(v8bf*)(ldsA + rl * 32 + kl) =
          *(const v8bf*)(X + (size_t)(mb + rl) * DD + k0 + kl);
    }
#pragma unroll
    for (int it = 0; it < 4; ++it) {
      int e = (tid + it * 128) * 4;
      int rl = e >> 5, kl = e & 31;
      v4f wa = *(const v4f*)(W + (size_t)(nb + rl) * DD + k0 + kl);
      v4bf wb = {(__bf16)wa.x, (__bf16)wa.y, (__bf16)wa.z, (__bf16)wa.w};
      *(v4bf*)(ldsW + rl * 32 + kl) = wb;
    }
    __syncthreads();
    v16bf af = load_afrag(ldsA, 32, wave * 16 + m16, hi);
#pragma unroll
    for (int j = 0; j < 4; ++j) {
      v16bf bf = load_bfrag(ldsW, 32, j * 16 + m16, hi);
      acc[j] = wmma_bf16(af, bf, acc[j]);
    }
  }
#pragma unroll
  for (int j = 0; j < 4; ++j) {
    int n_g = nb + j * 16 + m16;
    float bv = bias[n_g];
#pragma unroll
    for (int r = 0; r < 8; ++r) {
      int m_g = mb + wave * 16 + r + hi * 8;
      Y[(size_t)m_g * DD + n_g] = acc[j][r] + bv;
    }
  }
}

// ---------------------------------------------------------------------------
// Hopfield flash attention: for each (b,h,64-row q-tile) do
//   xi = q; repeat 3x: xi = softmax_causal(beta * xi K^T) K; out = softmax(...) V
// block = 128 threads (4 waves x 16 rows); all passes fused (rows independent).
// K/V tiles staged with CDNA5 async global->LDS DMA (ASYNCcnt).
// ---------------------------------------------------------------------------
__global__ __launch_bounds__(128) void hopfield_attn(
    const __bf16* __restrict__ qws, const __bf16* __restrict__ kws,
    const __bf16* __restrict__ vws, __bf16* __restrict__ attn,
    const float* __restrict__ log_beta) {
  __shared__ __align__(32) __bf16 ldsK[64 * 64];   // K tile, row-major (s, d)
  __shared__ __align__(32) __bf16 ldsV[64 * 64];   // V tile, row-major (pass 3)
  __shared__ __align__(32) __bf16 ldsT[64 * 64];   // (K or V) tile transposed (d, s)
  __shared__ __align__(32) __bf16 ldsX[4 * 16 * 64]; // per-wave transpose scratch

  const int tid = threadIdx.x, wave = tid >> 5, lane = tid & 31;
  const int m16 = lane & 15, hi = lane >> 4;
  const int qt = blockIdx.x, h = blockIdx.y, b = blockIdx.z;
  const int qb = qt * 64;
  const size_t bh = (size_t)b * HH + h;
  const __bf16* Kb = kws + (bh * SS << 6);
  const __bf16* Vb = vws + (bh * SS << 6);
  const __bf16* Qb = qws + (((bh * LL) + qb) << 6);
  const float beta = __expf(log_beta[0]);
  __bf16* myX = ldsX + wave * 16 * 64;
  const int wrow0 = qb + wave * 16;
  const v8f vz = {0.f, 0.f, 0.f, 0.f, 0.f, 0.f, 0.f, 0.f};

  // xi = q (two 16x32 A-fragments per wave)
  v16bf xi0 = load_afrag(Qb + wave * 16 * 64, 64, m16, hi);
  v16bf xi1 = load_afrag(Qb + wave * 16 * 64 + 32, 64, m16, hi);

  for (int pass = 0; pass < 4; ++pass) {
    const bool usev = (pass == 3);
    float rmax[8], rsum[8];
#pragma unroll
    for (int r = 0; r < 8; ++r) { rmax[r] = -1e30f; rsum[r] = 0.f; }
    v8f acc[4] = {vz, vz, vz, vz};

    for (int st = 0; st <= qt; ++st) {           // causal: skip tiles past diag
      const int s0 = st * 64;
      __syncthreads();                           // protect prior-tile readers
      // ---- async DMA stage of row-major tiles (16B per lane per issue) ----
#pragma unroll
      for (int it = 0; it < 4; ++it) {
        int ch = tid + it * 128;                 // 512 chunks of 8 halves
        int sl = ch >> 3, d = (ch & 7) * 8;
        unsigned boff = (unsigned)((((s0 + sl) << 6) + d) * 2);
        async_g2l_b128(ldsK + sl * 64 + d, Kb, boff);
        if (usev) async_g2l_b128(ldsV + sl * 64 + d, Vb, boff);
      }
      wait_async0();
      __syncthreads();                           // all waves' DMAs landed
      // ---- build transposed copy from LDS; prefetch next s-tile ----
      const __bf16* Tsrc = usev ? ldsV : ldsK;
#pragma unroll
      for (int it = 0; it < 4; ++it) {
        int ch = tid + it * 128;
        int sl = ch >> 3, d = (ch & 7) * 8;
        v8bf tv = *(const v8bf*)(Tsrc + sl * 64 + d);
#pragma unroll
        for (int j = 0; j < 8; ++j) ldsT[(d + j) * 64 + sl] = tv[j];
        if (st < qt)
          __builtin_prefetch(Kb + (((size_t)(s0 + 64 + sl)) << 6) + d, 0, 3);
      }
      __syncthreads();

#pragma unroll
      for (int c = 0; c < 2; ++c) {
        const int sc = s0 + c * 32;
        if (sc > wrow0) continue;                // chunk fully masked for wave

        // ---- scores = beta * xi K^T (two 16x16 tiles over 32 s-cols) ----
        v8f sco[2];
#pragma unroll
        for (int t = 0; t < 2; ++t) {
          int sl = c * 32 + t * 16 + m16;        // B row = key index in tile
          v8f s = vz;
          s = wmma_bf16(xi0, load_bfrag(ldsK, 64, sl, hi), s);
          s = wmma_bf16(xi1, load_bfrag(ldsK + 32, 64, sl, hi), s);
          sco[t] = s;
        }
        // ---- causal mask + online softmax update ----
#pragma unroll
        for (int r = 0; r < 8; ++r) {
          int rowg = wrow0 + r + hi * 8;
          float s0v = sco[0][r] * beta; if (sc + m16 > rowg)      s0v = -1e30f;
          float s1v = sco[1][r] * beta; if (sc + 16 + m16 > rowg) s1v = -1e30f;
          float mx = fmaxf(s0v, s1v);
          mx = fmaxf(mx, __shfl_xor(mx, 1));
          mx = fmaxf(mx, __shfl_xor(mx, 2));
          mx = fmaxf(mx, __shfl_xor(mx, 4));
          mx = fmaxf(mx, __shfl_xor(mx, 8));
          mx = fmaxf(mx, rmax[r]);
          float scale = __expf(rmax[r] - mx);
          float p0 = __expf(s0v - mx), p1 = __expf(s1v - mx);
          float ps = p0 + p1;
          ps += __shfl_xor(ps, 1); ps += __shfl_xor(ps, 2);
          ps += __shfl_xor(ps, 4); ps += __shfl_xor(ps, 8);
          rsum[r] = rsum[r] * scale + ps;
          rmax[r] = mx;
#pragma unroll
          for (int j = 0; j < 4; ++j) acc[j][r] *= scale;
          int rr = r + hi * 8;                   // C-layout -> LDS (transpose)
          myX[rr * 64 + m16]      = (__bf16)p0;
          myX[rr * 64 + 16 + m16] = (__bf16)p1;
        }
        // ---- acc += P * (K|V) tile ----
        v16bf pf = load_afrag(myX, 64, m16, hi);
#pragma unroll
        for (int j = 0; j < 4; ++j) {
          v16bf bt = load_bfrag(ldsT + (sc - s0), 64, j * 16 + m16, hi);
          acc[j] = wmma_bf16(pf, bt, acc[j]);
        }
      }
    }

    if (pass < 3) {  // xi <- acc/rsum, C-layout -> A-fragments via LDS
#pragma unroll
      for (int j = 0; j < 4; ++j)
#pragma unroll
        for (int r = 0; r < 8; ++r) {
          int rr = r + hi * 8;
          myX[rr * 64 + j * 16 + m16] = (__bf16)(acc[j][r] / rsum[r]);
        }
      xi0 = load_afrag(myX, 64, m16, hi);
      xi1 = load_afrag(myX + 32, 64, m16, hi);
    } else {         // write attention output (B, L, D) bf16
#pragma unroll
      for (int j = 0; j < 4; ++j)
#pragma unroll
        for (int r = 0; r < 8; ++r) {
          int rr = r + hi * 8;
          int lg = qb + wave * 16 + rr;
          int col = h * 64 + j * 16 + m16;
          attn[((size_t)b * LL + lg) * DD + col] = (__bf16)(acc[j][r] / rsum[r]);
        }
    }
  }
}

// ---------------------------------------------------------------------------
extern "C" void kernel_launch(void* const* d_in, const int* in_sizes, int n_in,
                              void* d_out, int out_size, void* d_ws, size_t ws_size,
                              hipStream_t stream) {
  (void)in_sizes; (void)n_in; (void)out_size; (void)ws_size;
  const float* query = (const float*)d_in[0];
  const float* key_  = (const float*)d_in[1];
  const float* value = (const float*)d_in[2];
  // d_in[3]: causal mask -- computed analytically in-kernel
  const float* Wq = (const float*)d_in[4];
  const float* bq = (const float*)d_in[5];
  const float* Wk = (const float*)d_in[6];
  const float* bk = (const float*)d_in[7];
  const float* Wv = (const float*)d_in[8];
  const float* bv = (const float*)d_in[9];
  const float* Wo = (const float*)d_in[10];
  const float* bo = (const float*)d_in[11];
  const float* lb = (const float*)d_in[12];

  char* ws = (char*)d_ws;
  __bf16* qws = (__bf16*)(ws);
  __bf16* kws = (__bf16*)(ws + (size_t)8 * 1024 * 1024);
  __bf16* vws = (__bf16*)(ws + (size_t)16 * 1024 * 1024);
  __bf16* aws = (__bf16*)(ws + (size_t)24 * 1024 * 1024);

  dim3 blk(128);
  dim3 gP(128, 8);          // M=8192/64, N=512/64
  proj_kernel<<<gP, blk, 0, stream>>>(query, Wq, bq, qws);
  proj_kernel<<<gP, blk, 0, stream>>>(key_,  Wk, bk, kws);
  proj_kernel<<<gP, blk, 0, stream>>>(value, Wv, bv, vws);

  dim3 gA(LL / 64, HH, 4);  // (q-tiles, heads, batch)
  hopfield_attn<<<gA, blk, 0, stream>>>(qws, kws, vws, aws, lb);

  oproj_kernel<<<gP, blk, 0, stream>>>(aws, Wo, bo, (float*)d_out);
}
